// MaskedResidualBlock_55233279426898
// MI455X (gfx1250) — compile-verified, matrix operand order checked by reading, not run on hardware
//
#include <hip/hip_runtime.h>

// MaskedResidualBlock for MI455X (gfx1250, wave32).
// Phase 1 (memory-bound, ~140MB stream): hypernetwork param-gen on VALU,
//   masked weights stored TRANSPOSED as bf16 for WMMA-friendly B loads.
// Phase 2 (compute-heavy, 17.2 GFLOP): two fused 256x512 @ 512x512 matmuls
//   per batch using v_wmma_f32_16x16x32_bf16, async global->LDS weight
//   staging (double buffered, ASYNCcnt), intermediate kept in LDS as bf16.

#define FDIM  512
#define BATCH 64
#define NSEQ  256
#define CDIM  128

typedef __attribute__((ext_vector_type(16))) __bf16 v16bf;
typedef __attribute__((ext_vector_type(8)))  float  v8f;
typedef __attribute__((ext_vector_type(4)))  int    v4i;

typedef __attribute__((address_space(1))) v4i* gas_v4i_ptr;  // global
typedef __attribute__((address_space(3))) v4i* las_v4i_ptr;  // LDS

union Frag {
  uint4  u[2];
  v16bf  v;
};

static __device__ __forceinline__ unsigned short f2bf(float f) {
  unsigned int u = __builtin_bit_cast(unsigned int, f);
  u += 0x7FFFu + ((u >> 16) & 1u);  // round-to-nearest-even
  return (unsigned short)(u >> 16);
}

#if __has_builtin(__builtin_amdgcn_global_load_async_to_lds_b128)
#define HAVE_ASYNC_LDS 1
#endif

// ---------------------------------------------------------------------------
// Phase 1: params[b,p] = phi[b,:] . Wc[p,:] + bc[p]; mask; store bf16 wT[b][k][i]
// ---------------------------------------------------------------------------
static __device__ __forceinline__ void dot_all_batches(
    const float* __restrict__ wrow, const float* sphi, float* acc) {
#pragma unroll
  for (int b = 0; b < BATCH; ++b) acc[b] = 0.f;
  for (int c4 = 0; c4 < CDIM; c4 += 4) {
    float4 w4 = *(const float4*)(wrow + c4);
#pragma unroll
    for (int b = 0; b < BATCH; ++b) {
      const float* pb = sphi + b * CDIM + c4;
      acc[b] = fmaf(w4.x, pb[0], acc[b]);
      acc[b] = fmaf(w4.y, pb[1], acc[b]);
      acc[b] = fmaf(w4.z, pb[2], acc[b]);
      acc[b] = fmaf(w4.w, pb[3], acc[b]);
    }
  }
}

__global__ void gen_weights(const float* __restrict__ phi,
                            const float* __restrict__ Wc,
                            const float* __restrict__ bc,
                            unsigned short* __restrict__ wT,   // [B][512][512] bf16 (transposed: [k][i])
                            float* __restrict__ biasOut) {     // [B][512] f32
  __shared__ float sphi[BATCH * CDIM];  // 32 KB
  for (int idx = threadIdx.x; idx < BATCH * CDIM; idx += 256) sphi[idx] = phi[idx];
  __syncthreads();

  float acc[BATCH];
  if (blockIdx.x < 1024) {
    int g = blockIdx.x * 256 + threadIdx.x;  // enumerate (k major, i minor) for coalesced stores
    int i = g & (FDIM - 1);                  // weight row (reduction dim of hidden matmul)
    int k = g >> 9;                          // weight col (output dim)
    bool live = (i % 63) >= (k % 63);        // mask[i,k] = d[i] >= d[k], d = x%63+1
    if (live) {
      int p = i * FDIM + k;
      dot_all_batches(Wc + (size_t)p * CDIM, sphi, acc);
      float bias = bc[p];
#pragma unroll
      for (int b = 0; b < BATCH; ++b)
        wT[(size_t)b * FDIM * FDIM + (size_t)k * FDIM + i] = f2bf(acc[b] + bias);
    } else {
      // masked: skip the 512B Wc row entirely (saves ~half of the 134MB stream)
#pragma unroll
      for (int b = 0; b < BATCH; ++b)
        wT[(size_t)b * FDIM * FDIM + (size_t)k * FDIM + i] = 0;
    }
  } else {
    int k = (int)(blockIdx.x - 1024) * 256 + threadIdx.x;  // 0..511
    int p = FDIM * FDIM + k;
    dot_all_batches(Wc + (size_t)p * CDIM, sphi, acc);
    float bias = bc[p];
#pragma unroll
    for (int b = 0; b < BATCH; ++b)
      biasOut[(size_t)b * FDIM + k] = acc[b] + bias;
  }
}

// ---------------------------------------------------------------------------
// Phase 2: out = x + (relu(relu(x)@mw + bias) @ mw + bias)
// One workgroup per (batch, 64-row tile); 8 waves; WMMA bf16->f32.
// ---------------------------------------------------------------------------
#define AROW 520        // shA row stride in bf16 elems (1040B: conflict-free A frags)
#define WPAD 40         // shW row stride in bf16 elems (80B: conflict-free B frags)

__global__ void __launch_bounds__(256, 2) fused_block(
    const float* __restrict__ inp,            // [B][256][512]
    const unsigned short* __restrict__ wT,    // [B][512][512] bf16 transposed
    const float* __restrict__ biasW,          // [B][512]
    float* __restrict__ out) {
  __shared__ unsigned short shA[64][AROW];        // 66,560 B  (A tile, reused for t1)
  __shared__ unsigned short shW[2][FDIM][WPAD];   // 81,920 B  (double-buffered weight K-slices)
  __shared__ float shBias[FDIM];                  //  2,048 B

  const int tid   = threadIdx.x;
  const int lane  = tid & 31;
  const int wv    = tid >> 5;
  const int m     = wv & 3;                // 16-row tile within the 64-row block
  const int cbase = (wv >> 2) * 256;       // 16 column tiles of 16 each
  const int b     = blockIdx.y;
  const int n0    = blockIdx.x * 64;

  const unsigned short* wB = wT + (size_t)b * FDIM * FDIM;
  const float* xB = inp + ((size_t)b * NSEQ + n0) * FDIM;

  // Stage relu(X) as bf16 into LDS; stage bias.
  for (int c = tid; c < 64 * 128; c += 256) {
    int r = c >> 7, c4 = (c & 127) * 4;
    float4 x = *(const float4*)(xB + (size_t)r * FDIM + c4);
    shA[r][c4 + 0] = f2bf(fmaxf(x.x, 0.f));
    shA[r][c4 + 1] = f2bf(fmaxf(x.y, 0.f));
    shA[r][c4 + 2] = f2bf(fmaxf(x.z, 0.f));
    shA[r][c4 + 3] = f2bf(fmaxf(x.w, 0.f));
  }
  for (int c = tid; c < FDIM; c += 256) shBias[c] = biasW[(size_t)b * FDIM + c];

  // Async stage one 32(i) x 512(j) weight slice (wT rows j, 64B each) into LDS.
  auto prefetch = [&](int ks, int bufi) {
#pragma unroll
    for (int u = 0; u < 8; ++u) {
      int q = u * 256 + tid;     // 2048 x 16B chunks
      int j = q >> 2, seg = q & 3;
      const unsigned short* g = wB + (size_t)j * FDIM + ks * 32 + seg * 8;
      unsigned short* l = &shW[bufi][j][seg * 8];
#ifdef HAVE_ASYNC_LDS
      __builtin_amdgcn_global_load_async_to_lds_b128(
          (gas_v4i_ptr)(void*)g, (las_v4i_ptr)(void*)l, 0, 0);
#else
      *(uint4*)l = *(const uint4*)g;
#endif
    }
  };
  auto wait_async = [&]() {
#ifdef HAVE_ASYNC_LDS
#if __has_builtin(__builtin_amdgcn_s_wait_asynccnt)
    __builtin_amdgcn_s_wait_asynccnt(0);
#else
    asm volatile("s_wait_asynccnt 0" ::: "memory");
#endif
#endif
  };

  const int sel = lane >> 4;     // which K-half of the fragment this lane holds
  const int lr  = lane & 15;     // A: row within 16-tile; B: column within 16-tile

  auto matmul = [&](v8f (&acc)[16]) {
    prefetch(0, 0);
    wait_async();
    __syncthreads();
    for (int ks = 0; ks < 16; ++ks) {          // K = 512 in steps of 32
      if (ks < 15) prefetch(ks + 1, (ks + 1) & 1);
      // A fragment (16x32 bf16): lane lr holds K = ks*32 + {sel*8..+8, 16+sel*8..+8}
      const unsigned short* ap = &shA[m * 16 + lr][ks * 32 + sel * 8];
      Frag A;
      A.u[0] = *(const uint4*)ap;
      A.u[1] = *(const uint4*)(ap + 16);
      const unsigned short(*wS)[WPAD] = shW[ks & 1];
#pragma unroll
      for (int t = 0; t < 16; ++t) {
        // B fragment (32x16): lane lr = column j, contiguous i runs from transposed wT
        const unsigned short* bp = &wS[cbase + t * 16 + lr][sel * 8];
        Frag Bv;
        Bv.u[0] = *(const uint4*)bp;
        Bv.u[1] = *(const uint4*)(bp + 16);
        acc[t] = __builtin_amdgcn_wmma_f32_16x16x32_bf16(
            false, A.v, false, Bv.v, (short)0, acc[t], false, false);
      }
      wait_async();
      __syncthreads();   // slice ks+1 visible to all; everyone done reading slice ks
    }
  };

  const int col = lane & 15, half = lane >> 4;  // C/D layout: col=lane%16, rows v + 8*half
  v8f zero = {};
  v8f acc[16];
#pragma unroll
  for (int t = 0; t < 16; ++t) acc[t] = zero;

  // ---- matmul 1: t1 = relu(relu(x) @ mw + bias) ----
  matmul(acc);
#pragma unroll
  for (int t = 0; t < 16; ++t) {
    int j0 = cbase + t * 16;
    float bb = shBias[j0 + col];
#pragma unroll
    for (int v = 0; v < 8; ++v) {
      float f = acc[t][v] + bb;
      shA[m * 16 + half * 8 + v][j0 + col] = f2bf(fmaxf(f, 0.f));  // overwrite A tile with t1
    }
  }
  __syncthreads();

  // ---- matmul 2: out = x + t1 @ mw + bias ----
#pragma unroll
  for (int t = 0; t < 16; ++t) acc[t] = zero;
  matmul(acc);
#pragma unroll
  for (int t = 0; t < 16; ++t) {
    int j0 = cbase + t * 16;
    float bb = shBias[j0 + col];
#pragma unroll
    for (int v = 0; v < 8; ++v) {
      int n = m * 16 + half * 8 + v;
      size_t off = ((size_t)b * NSEQ + n0 + n) * FDIM + j0 + col;
      out[off] = inp[off] + acc[t][v] + bb;
    }
  }
}

// ---------------------------------------------------------------------------
extern "C" void kernel_launch(void* const* d_in, const int* in_sizes, int n_in,
                              void* d_out, int out_size, void* d_ws, size_t ws_size,
                              hipStream_t stream) {
  const float* inp = (const float*)d_in[0];   // (1,64,256,512)
  const float* phi = (const float*)d_in[1];   // (1,64,128)
  const float* Wc  = (const float*)d_in[2];   // (262656,128)
  const float* bc  = (const float*)d_in[3];   // (262656,)
  float* out = (float*)d_out;

  unsigned short* wT = (unsigned short*)d_ws;                       // 32 MB bf16
  float* biasW = (float*)((char*)d_ws +
                          (size_t)BATCH * FDIM * FDIM * sizeof(unsigned short));

  gen_weights<<<1026, 256, 0, stream>>>(phi, Wc, bc, wT, biasW);
  fused_block<<<dim3(NSEQ / 64, BATCH), 256, 0, stream>>>(inp, wT, biasW, out);
}
